// StatementClassifier_38147899523813
// MI455X (gfx1250) — compile-verified, matrix operand order checked by reading, not run on hardware
//
#include <hip/hip_runtime.h>
#include <math.h>

// ---------------- WMMA types ----------------
typedef __attribute__((ext_vector_type(16))) _Float16 v16h;
typedef __attribute__((ext_vector_type(8)))  float    v8f;

// ---------------- problem constants ----------------
#define G_      8192
#define N_AST_  32
#define N0_     (G_ * N_AST_)          // 262144 AST nodes
#define E_AST_  (G_ * (N_AST_ - 1))    // 253952 AST edges
#define K0_     23                     // ceil(0.7*32)
#define N1_     (G_ * K0_)             // 188416
#define K1_     12                     // ceil(0.5*23)
#define N_STMT_ 8192
#define E_STMT_ 32768
#define N_SEL_  4096

// =====================================================================
// Block-cooperative WMMA GEMM: C[M,N] = act(A[M,K] @ B[K,N] + bias)
// 256-thread block (8 waves) computes a 128x64 macro-tile.
//  - B k-slab (32x64) staged in LDS, f32->f16, pre-swizzled into the
//    ISA 16-bit WMMA B-fragment layout (v16h per lane per 16x16 tile).
//  - each wave holds one A fragment (from global, 2x float4 runs) and
//    issues 4 WMMAs per k-step (4 N-tiles), 8x B reuse across waves.
//  - global_prefetch_b8 for the next k-slab via __builtin_prefetch.
// Requires M%128==0, N%64==0, K%32==0 (true for every call site).
// act: 0=none, 1=relu.
// =====================================================================
__global__ void k_gemm_wmma(const float* __restrict__ A,
                            const float* __restrict__ B,
                            const float* __restrict__ bias,
                            float* __restrict__ C,
                            int M, int N, int K, int act, int nblocks)
{
  __shared__ v16h lds_b[4][32];              // 4 N-tiles x 32 lanes x 16 halves = 4KB

  const int wave = threadIdx.x >> 5;
  const int lane = threadIdx.x & 31;
  const int mb = blockIdx.x / nblocks;
  const int nb = blockIdx.x % nblocks;
  const int nblock = nb * 64;

  const int row_a = mb * 128 + wave * 16 + (lane & 15);
  const int khalf = (lane >= 16) ? 8 : 0;

  // producer coordinates for the B slab: thread t loads 8 consecutive
  // floats of row k = t>>3 starting at n = (t&7)*8
  const int pk   = threadIdx.x >> 3;         // 0..31
  const int pn0  = (threadIdx.x & 7) * 8;    // 0..56
  const int plhi = (pk >> 3) & 1;            // target lane group (0: lanes0-15)
  const int pe   = (pk & 7) + ((pk >> 4) << 3);  // target element within v16h

  v8f acc0 = {}, acc1 = {}, acc2 = {}, acc3 = {};

  for (int kk0 = 0; kk0 < K; kk0 += 32) {
    // ---- load B slab (f32) to registers ----
    const float* bp = B + (size_t)(kk0 + pk) * N + nblock + pn0;
    float bf[8];
    *(float4*)(bf + 0) = *(const float4*)(bp + 0);
    *(float4*)(bf + 4) = *(const float4*)(bp + 4);

    __syncthreads();                         // previous readers done
    // ---- store into LDS in WMMA B-fragment layout (f16) ----
#pragma unroll
    for (int i = 0; i < 8; ++i) {
      int n = pn0 + i;
      ((_Float16*)&lds_b[n >> 4][(n & 15) + 16 * plhi])[pe] = (_Float16)bf[i];
    }

    // ---- A fragment from global (two 8-float runs per lane) ----
    const float* ap = A + (size_t)row_a * K + kk0 + khalf;
    float af[16];
    *(float4*)(af + 0)  = *(const float4*)(ap + 0);
    *(float4*)(af + 4)  = *(const float4*)(ap + 4);
    *(float4*)(af + 8)  = *(const float4*)(ap + 16);
    *(float4*)(af + 12) = *(const float4*)(ap + 20);
    v16h a;
#pragma unroll
    for (int e = 0; e < 16; ++e) a[e] = (_Float16)af[e];

    // prefetch next k-slab (lowers to global_prefetch_b8 on gfx1250)
    if (kk0 + 32 < K) {
      __builtin_prefetch(ap + 32, 0, 1);
      __builtin_prefetch(B + (size_t)(kk0 + 32 + pk) * N + nblock + pn0, 0, 1);
    }

    __syncthreads();                         // LDS slab ready
    v16h b0 = lds_b[0][lane];
    v16h b1 = lds_b[1][lane];
    v16h b2 = lds_b[2][lane];
    v16h b3 = lds_b[3][lane];
    acc0 = __builtin_amdgcn_wmma_f32_16x16x32_f16(false, a, false, b0, (short)0, acc0, false, false);
    acc1 = __builtin_amdgcn_wmma_f32_16x16x32_f16(false, a, false, b1, (short)0, acc1, false, false);
    acc2 = __builtin_amdgcn_wmma_f32_16x16x32_f16(false, a, false, b2, (short)0, acc2, false, false);
    acc3 = __builtin_amdgcn_wmma_f32_16x16x32_f16(false, a, false, b3, (short)0, acc3, false, false);
  }

  // ---- epilogue: C/D layout (lanes0-15: M=v, lanes16-31: M=8+v) ----
  const int mrow = mb * 128 + wave * 16 + ((lane >= 16) ? 8 : 0);
  const int colq = lane & 15;
#pragma unroll
  for (int nt = 0; nt < 4; ++nt) {
    v8f acc = (nt == 0) ? acc0 : (nt == 1) ? acc1 : (nt == 2) ? acc2 : acc3;
    int col = nblock + nt * 16 + colq;
    float bb = bias ? bias[col] : 0.0f;
#pragma unroll
    for (int v = 0; v < 8; ++v) {
      float x = acc[v] + bb;
      if (act == 1) x = fmaxf(x, 0.0f);
      C[(size_t)(mrow + v) * N + col] = x;
    }
  }
}

// ---------------- small utility kernels ----------------
__global__ void k_fill(float* p, float val, long long n)
{
  long long i = (long long)blockIdx.x * blockDim.x + threadIdx.x;
  if (i < n) p[i] = val;
}

__global__ void k_init_bias(float* out, const float* __restrict__ bias, int n, int D)
{
  long long i = (long long)blockIdx.x * blockDim.x + threadIdx.x;
  if (i < (long long)n * D) out[i] = bias[i % D];
}

__device__ inline void atomicMaxF(float* addr, float val)
{
  int* ia = (int*)addr;
  int cur = *ia;
  while (__int_as_float(cur) < val) {
    int prev = atomicCAS(ia, cur, __float_as_int(val));
    if (prev == cur) break;
    cur = prev;
  }
}

// ---------------- GAT attention ----------------
__global__ void k_attn_node(const float* __restrict__ h,
                            const float* __restrict__ a_src,
                            const float* __restrict__ a_dst,
                            float* asrc, float* adst, int n, int heads, int C)
{
  int idx = blockIdx.x * blockDim.x + threadIdx.x;
  if (idx >= n * heads) return;
  int node = idx / heads, hh = idx % heads;
  const float* hp = h + (size_t)node * heads * C + (size_t)hh * C;
  float s1 = 0.f, s2 = 0.f;
  for (int c = 0; c < C; ++c) {
    float v = hp[c];
    s1 += v * a_src[hh * C + c];
    s2 += v * a_dst[hh * C + c];
  }
  asrc[idx] = s1;
  adst[idx] = s2;
}

// edges [0,E) real, [E,E+nself) self loops. logits + atomic segment max.
__global__ void k_edge_logits(const int* __restrict__ src,
                              const int* __restrict__ dst,
                              const char* __restrict__ mask,
                              const float* __restrict__ asrc,
                              const float* __restrict__ adst,
                              float* logits, float* nmax,
                              int E, int nself, int heads)
{
  int idx = blockIdx.x * blockDim.x + threadIdx.x;
  if (idx >= (E + nself) * heads) return;
  int e = idx / heads, hh = idx % heads;
  int s, d; bool valid;
  if (e < E) { s = src[e]; d = dst[e]; valid = mask ? (mask[e] != 0) : true; }
  else       { s = d = e - E; valid = true; }
  float l = asrc[s * heads + hh] + adst[d * heads + hh];
  l = (l > 0.f) ? l : 0.2f * l;          // leaky_relu(0.2)
  if (!valid) l = -1e9f;
  logits[idx] = l;
  atomicMaxF(&nmax[d * heads + hh], l);
}

__global__ void k_edge_expsum(const int* __restrict__ dst,
                              float* logits,
                              const float* __restrict__ nmax,
                              float* nsum, int E, int nself, int heads)
{
  int idx = blockIdx.x * blockDim.x + threadIdx.x;
  if (idx >= (E + nself) * heads) return;
  int e = idx / heads, hh = idx % heads;
  int d = (e < E) ? dst[e] : (e - E);
  float ex = expf(logits[idx] - nmax[d * heads + hh]);
  logits[idx] = ex;
  atomicAdd(&nsum[d * heads + hh], ex);
}

__global__ void k_edge_scatter(const int* __restrict__ src,
                               const int* __restrict__ dst,
                               const char* __restrict__ mask,
                               const float* __restrict__ expe,
                               const float* __restrict__ nsum,
                               const float* __restrict__ h,
                               float* out, int E, int nself, int heads, int C)
{
  int idx = blockIdx.x * blockDim.x + threadIdx.x;
  if (idx >= (E + nself) * heads) return;
  int e = idx / heads, hh = idx % heads;
  int s, d; bool valid;
  if (e < E) { s = src[e]; d = dst[e]; valid = mask ? (mask[e] != 0) : true; }
  else       { s = d = e - E; valid = true; }
  if (!valid) return;
  float alpha = expe[idx] / fmaxf(nsum[d * heads + hh], 1e-16f);
  const float* hs = h + (size_t)s * heads * C + (size_t)hh * C;
  float* od = out + (size_t)d * heads * C + (size_t)hh * C;
  for (int c = 0; c < C; ++c) atomicAdd(&od[c], alpha * hs[c]);
}

// ---------------- BatchNorm (training-mode batch stats, biased var) ----
__global__ void k_bn_stats(const float* __restrict__ x, float* mean, float* var,
                           int n, int D, int do_tanh)
{
  int d = blockIdx.x;
  float s = 0.f, s2 = 0.f;
  for (int i = threadIdx.x; i < n; i += blockDim.x) {
    float v = x[(size_t)i * D + d];
    if (do_tanh) v = tanhf(v);
    s += v; s2 += v * v;
  }
  __shared__ float sh[256], sh2[256];
  int t = threadIdx.x;
  sh[t] = s; sh2[t] = s2;
  __syncthreads();
  for (int w = 128; w > 0; w >>= 1) {
    if (t < w) { sh[t] += sh[t + w]; sh2[t] += sh2[t + w]; }
    __syncthreads();
  }
  if (t == 0) {
    float mu = sh[0] / (float)n;
    mean[d] = mu;
    var[d] = sh2[0] / (float)n - mu * mu;
  }
}

__global__ void k_bn_apply(const float* __restrict__ x,
                           const float* __restrict__ mean,
                           const float* __restrict__ var,
                           const float* __restrict__ g,
                           const float* __restrict__ b,
                           float* y, int n, int D, int do_tanh)
{
  long long i = (long long)blockIdx.x * blockDim.x + threadIdx.x;
  if (i >= (long long)n * D) return;
  int d = (int)(i % D);
  float v = x[i];
  if (do_tanh) v = tanhf(v);
  y[i] = (v - mean[d]) * rsqrtf(var[d] + 1e-5f) * g[d] + b[d];
}

// ---------------- TopK pooling ----------------
__global__ void k_wnorm(const float* __restrict__ w, int D, float* out)
{
  __shared__ float sh[256];
  float s = 0.f;
  for (int i = threadIdx.x; i < D; i += blockDim.x) s += w[i] * w[i];
  sh[threadIdx.x] = s;
  __syncthreads();
  for (int o2 = 128; o2 > 0; o2 >>= 1) {
    if (threadIdx.x < o2) sh[threadIdx.x] += sh[threadIdx.x + o2];
    __syncthreads();
  }
  if (threadIdx.x == 0) out[0] = sqrtf(sh[0]);
}

__global__ void k_score(const float* __restrict__ x, const float* __restrict__ w,
                        const float* __restrict__ wnorm, float* score, int n, int D)
{
  int i = blockIdx.x * blockDim.x + threadIdx.x;
  if (i >= n) return;
  const float* xp = x + (size_t)i * D;
  float s = 0.f;
  for (int c = 0; c < D; ++c) s += xp[c] * w[c];
  score[i] = s / wnorm[0];
}

// one 32-thread block per graph; O(n^2) rank = descending order, ties by index
__global__ void k_topk(const float* __restrict__ x, const float* __restrict__ score,
                       float* new_x, int* new_id,
                       int n_per_g, int k, int D, int relu_flag)
{
  int g = blockIdx.x;
  int t = threadIdx.x;
  __shared__ float sc[32];
  if (t < n_per_g) sc[t] = score[g * n_per_g + t];
  __syncthreads();
  if (t >= n_per_g) return;
  float mys = sc[t];
  int rank = 0;
  for (int j = 0; j < n_per_g; ++j) {
    float sj = sc[j];
    if (sj > mys || (sj == mys && j < t)) ++rank;
  }
  int old = g * n_per_g + t;
  if (rank < k) {
    int nn = g * k + rank;
    new_id[old] = nn;
    float gate = tanhf(mys);
    const float* xp = x + (size_t)old * D;
    float* op = new_x + (size_t)nn * D;
    for (int c = 0; c < D; ++c) {
      float v = xp[c] * gate;
      if (relu_flag) v = fmaxf(v, 0.f);
      op[c] = v;
    }
  } else {
    new_id[old] = -1;
  }
}

__global__ void k_remap(const int* __restrict__ src, const int* __restrict__ dst,
                        const char* __restrict__ mask, const int* __restrict__ new_id,
                        int* nsrc, int* ndst, char* nmask, int E)
{
  int e = blockIdx.x * blockDim.x + threadIdx.x;
  if (e >= E) return;
  int ns = new_id[src[e]], nd = new_id[dst[e]];
  bool v = (mask ? (mask[e] != 0) : true) && ns >= 0 && nd >= 0;
  nsrc[e] = ns < 0 ? 0 : ns;
  ndst[e] = nd < 0 ? 0 : nd;
  nmask[e] = v ? 1 : 0;
}

__global__ void k_maxpool(const float* __restrict__ x, float* out, int ng, int kk, int D)
{
  int i = blockIdx.x * blockDim.x + threadIdx.x;
  if (i >= ng * D) return;
  int g = i / D, c = i % D;
  float m = -3.0e38f;
  for (int r = 0; r < kk; ++r)
    m = fmaxf(m, x[((size_t)g * kk + r) * D + c]);
  out[i] = fmaxf(m, 0.f);
}

// ---------------- RGAT ----------------
__global__ void k_rgat_logits(const int* __restrict__ src, const int* __restrict__ dst,
                              const int* __restrict__ etype,
                              const float* __restrict__ h0, const float* __restrict__ h1,
                              const float* __restrict__ q, const float* __restrict__ katt,
                              float* logits, float* nmax, int E, int heads, int D)
{
  int idx = blockIdx.x * blockDim.x + threadIdx.x;
  if (idx >= E * heads) return;
  int e = idx / heads, hh = idx % heads;
  const float* base = (etype[e] == 1) ? h1 : h0;
  const float* hj = base + (size_t)src[e] * D;
  const float* hi = base + (size_t)dst[e] * D;
  float l = 0.f;
  for (int c = 0; c < D; ++c)
    l += hi[c] * q[c * heads + hh] + hj[c] * katt[c * heads + hh];
  l = (l > 0.f) ? l : 0.2f * l;
  logits[idx] = l;
  atomicMaxF(&nmax[dst[e] * heads + hh], l);
}

__global__ void k_rgat_scatter(const int* __restrict__ src, const int* __restrict__ dst,
                               const int* __restrict__ etype,
                               const float* __restrict__ expe, const float* __restrict__ nsum,
                               const float* __restrict__ h0, const float* __restrict__ h1,
                               float* out, int E, int heads, int C)
{
  int idx = blockIdx.x * blockDim.x + threadIdx.x;
  if (idx >= E * heads) return;
  int e = idx / heads, hh = idx % heads;
  const float* base = (etype[e] == 1) ? h1 : h0;
  const float* hj = base + (size_t)src[e] * heads * C + (size_t)hh * C;
  int d = dst[e];
  float alpha = expe[idx] / fmaxf(nsum[d * heads + hh], 1e-16f);
  float* od = out + (size_t)d * heads * C + (size_t)hh * C;
  for (int c = 0; c < C; ++c) atomicAdd(&od[c], alpha * hj[c]);
}

// ---------------- tail ----------------
__global__ void k_gather(const float* __restrict__ x, const int* __restrict__ idxs,
                         float* y, int nsel, int D)
{
  int i = blockIdx.x * blockDim.x + threadIdx.x;
  if (i >= nsel * D) return;
  int r = i / D, c = i % D;
  y[i] = x[(size_t)idxs[r] * D + c];
}

__global__ void k_clf_final(const float* __restrict__ h, const float* __restrict__ w2,
                            const float* __restrict__ b2, float* out, int n, int D)
{
  int i = blockIdx.x * blockDim.x + threadIdx.x;
  if (i >= n) return;
  const float* hp = h + (size_t)i * D;
  float s = b2[0];
  for (int c = 0; c < D; ++c) s += hp[c] * w2[c];
  out[i] = 1.f / (1.f + expf(-s));
}

// =====================================================================
extern "C" void kernel_launch(void* const* d_in, const int* in_sizes, int n_in,
                              void* d_out, int out_size, void* d_ws, size_t ws_size,
                              hipStream_t stream)
{
  (void)in_sizes; (void)n_in; (void)out_size; (void)ws_size;

  // -------- inputs (setup_inputs order) --------
  const float* ast_x    = (const float*)d_in[0];
  const int*   ast_src  = (const int*)d_in[1];
  const int*   ast_dst  = (const int*)d_in[2];
  const int*   stmt_src = (const int*)d_in[3];
  const int*   stmt_dst = (const int*)d_in[4];
  const int*   stmt_et  = (const int*)d_in[5];
  const int*   sel_idx  = (const int*)d_in[6];
  const float* gat0_W   = (const float*)d_in[7];
  const float* gat0_as  = (const float*)d_in[8];
  const float* gat0_ad  = (const float*)d_in[9];
  const float* gat0_b   = (const float*)d_in[10];
  const float* bn0_g    = (const float*)d_in[11];
  const float* bn0_b    = (const float*)d_in[12];
  const float* pool0_w  = (const float*)d_in[13];
  const float* gat1_W   = (const float*)d_in[14];
  const float* gat1_as  = (const float*)d_in[15];
  const float* gat1_ad  = (const float*)d_in[16];
  const float* gat1_b   = (const float*)d_in[17];
  const float* bn1_g    = (const float*)d_in[18];
  const float* bn1_b    = (const float*)d_in[19];
  const float* pool1_w  = (const float*)d_in[20];
  const float* enc_w1   = (const float*)d_in[21];
  const float* enc_b1   = (const float*)d_in[22];
  const float* enc_w2   = (const float*)d_in[23];
  const float* enc_b2   = (const float*)d_in[24];
  const float* rgat0_W  = (const float*)d_in[25];
  const float* rgat0_q  = (const float*)d_in[26];
  const float* rgat0_k  = (const float*)d_in[27];
  const float* rgat0_b  = (const float*)d_in[28];
  const float* bnc0_g   = (const float*)d_in[29];
  const float* bnc0_b   = (const float*)d_in[30];
  const float* rgat1_W  = (const float*)d_in[31];
  const float* rgat1_q  = (const float*)d_in[32];
  const float* rgat1_k  = (const float*)d_in[33];
  const float* rgat1_b  = (const float*)d_in[34];
  const float* bnc1_g   = (const float*)d_in[35];
  const float* bnc1_b   = (const float*)d_in[36];
  const float* clf_w1   = (const float*)d_in[37];
  const float* clf_b1   = (const float*)d_in[38];
  const float* clf_w2   = (const float*)d_in[39];
  const float* clf_b2   = (const float*)d_in[40];
  float* out = (float*)d_out;

  // -------- workspace arena --------
  float* W = (float*)d_ws;
  size_t off = 0;
  auto F = [&](size_t n) { float* p = W + off; off += n; return p; };

  const size_t BIG = (size_t)N0_ * 192;       // 50,331,648 floats
  float* A0     = F(BIG);                     // ping node-feature buffer
  float* B0     = F(BIG);                     // pong node-feature buffer
  float* asrc   = F((size_t)N0_ * 3);
  float* adst   = F((size_t)N0_ * 3);
  float* elog   = F((size_t)(E_AST_ + N0_) * 3);
  float* nmax   = F((size_t)N0_ * 3);
  float* nsum   = F((size_t)N0_ * 3);
  float* bnmean = F(512);
  float* bnvar  = F(512);
  float* wnrm   = F(1);
  float* score  = F(N0_);
  int*   new_id = (int*)F(N0_);
  int*   es1    = (int*)F(E_AST_);
  int*   ed1    = (int*)F(E_AST_);
  char*  em1    = (char*)F(E_AST_ / 4 + 4);

  // statement-phase buffers overlay B0 (dead after the maxpool step)
  float* tmp1 = B0;                       // 8192x64
  float* xenc = tmp1 + (size_t)8192 * 64; // 8192x128
  float* h0s  = xenc + (size_t)8192 * 128;// 8192x384
  float* h1s  = h0s + (size_t)8192 * 384;
  float* outs = h1s + (size_t)8192 * 384;
  float* hbn  = outs + (size_t)8192 * 384;
  float* h0b  = hbn + (size_t)8192 * 384; // 8192x128
  float* h1b  = h0b + (size_t)8192 * 128;
  float* out2 = h1b + (size_t)8192 * 128;
  float* hf   = out2 + (size_t)8192 * 128;
  float* hsel = hf + (size_t)8192 * 128;  // 4096x128
  float* c1   = hsel + (size_t)4096 * 128;

  auto gemm = [&](const float* Aa, const float* Bb, const float* bias, float* Cc,
                  int M, int N, int K, int act) {
    int mblocks = M / 128, nblocks = N / 64;
    k_gemm_wmma<<<mblocks * nblocks, 256, 0, stream>>>(Aa, Bb, bias, Cc, M, N, K, act, nblocks);
  };
  auto grid1 = [&](long long n) { return (int)((n + 255) / 256); };

  // ================= StatementEncoder =================
  // GAT0: h = ast_x @ W  [N0,192]
  gemm(ast_x, gat0_W, nullptr, A0, N0_, 192, 128, 0);
  k_attn_node<<<grid1((long long)N0_ * 3), 256, 0, stream>>>(A0, gat0_as, gat0_ad, asrc, adst, N0_, 3, 64);
  k_fill<<<grid1((long long)N0_ * 3), 256, 0, stream>>>(nmax, -3.0e38f, (long long)N0_ * 3);
  k_fill<<<grid1((long long)N0_ * 3), 256, 0, stream>>>(nsum, 0.f, (long long)N0_ * 3);
  {
    long long tot = (long long)(E_AST_ + N0_) * 3;
    k_edge_logits<<<grid1(tot), 256, 0, stream>>>(ast_src, ast_dst, nullptr, asrc, adst, elog, nmax, E_AST_, N0_, 3);
    k_edge_expsum<<<grid1(tot), 256, 0, stream>>>(ast_dst, elog, nmax, nsum, E_AST_, N0_, 3);
    k_init_bias<<<grid1((long long)N0_ * 192), 256, 0, stream>>>(B0, gat0_b, N0_, 192);
    k_edge_scatter<<<grid1(tot), 256, 0, stream>>>(ast_src, ast_dst, nullptr, elog, nsum, A0, B0, E_AST_, N0_, 3, 64);
  }
  // BN0(tanh) -> A0
  k_bn_stats<<<192, 256, 0, stream>>>(B0, bnmean, bnvar, N0_, 192, 1);
  k_bn_apply<<<grid1((long long)N0_ * 192), 256, 0, stream>>>(B0, bnmean, bnvar, bn0_g, bn0_b, A0, N0_, 192, 1);
  // TopK0 (ratio 0.7 -> k=23), relu fused into gated features -> B0
  k_wnorm<<<1, 256, 0, stream>>>(pool0_w, 192, wnrm);
  k_score<<<grid1(N0_), 256, 0, stream>>>(A0, pool0_w, wnrm, score, N0_, 192);
  k_topk<<<G_, 32, 0, stream>>>(A0, score, B0, new_id, N_AST_, K0_, 192, 1);
  k_remap<<<grid1(E_AST_), 256, 0, stream>>>(ast_src, ast_dst, nullptr, new_id, es1, ed1, em1, E_AST_);

  // GAT1: h = new_x @ W  [N1,64]
  gemm(B0, gat1_W, nullptr, A0, N1_, 64, 192, 0);
  k_attn_node<<<grid1(N1_), 256, 0, stream>>>(A0, gat1_as, gat1_ad, asrc, adst, N1_, 1, 64);
  k_fill<<<grid1(N1_), 256, 0, stream>>>(nmax, -3.0e38f, N1_);
  k_fill<<<grid1(N1_), 256, 0, stream>>>(nsum, 0.f, N1_);
  {
    long long tot = (long long)(E_AST_ + N1_);
    k_edge_logits<<<grid1(tot), 256, 0, stream>>>(es1, ed1, em1, asrc, adst, elog, nmax, E_AST_, N1_, 1);
    k_edge_expsum<<<grid1(tot), 256, 0, stream>>>(ed1, elog, nmax, nsum, E_AST_, N1_, 1);
    k_init_bias<<<grid1((long long)N1_ * 64), 256, 0, stream>>>(B0, gat1_b, N1_, 64);
    k_edge_scatter<<<grid1(tot), 256, 0, stream>>>(es1, ed1, em1, elog, nsum, A0, B0, E_AST_, N1_, 1, 64);
  }
  // BN1(tanh) -> A0
  k_bn_stats<<<64, 256, 0, stream>>>(B0, bnmean, bnvar, N1_, 64, 1);
  k_bn_apply<<<grid1((long long)N1_ * 64), 256, 0, stream>>>(B0, bnmean, bnvar, bn1_g, bn1_b, A0, N1_, 64, 1);
  // TopK1 (ratio 0.5 on 23 -> k=12), no relu on gated features -> B0
  k_wnorm<<<1, 256, 0, stream>>>(pool1_w, 64, wnrm);
  k_score<<<grid1(N1_), 256, 0, stream>>>(A0, pool1_w, wnrm, score, N1_, 64);
  k_topk<<<G_, 32, 0, stream>>>(A0, score, B0, new_id, K0_, K1_, 64, 0);
  // global max pool + relu -> A0 [G,64]
  k_maxpool<<<grid1((long long)G_ * 64), 256, 0, stream>>>(B0, A0, G_, K1_, 64);
  // encoder MLP -> xenc [G,128]
  gemm(A0, enc_w1, enc_b1, tmp1, G_, 64, 64, 1);
  gemm(tmp1, enc_w2, enc_b2, xenc, G_, 128, 64, 0);

  // ================= StatementClassifier =================
  // RGAT0
  gemm(xenc, rgat0_W,             nullptr, h0s, N_STMT_, 384, 128, 0);
  gemm(xenc, rgat0_W + 128 * 384, nullptr, h1s, N_STMT_, 384, 128, 0);
  k_fill<<<grid1((long long)N_STMT_ * 3), 256, 0, stream>>>(nmax, -3.0e38f, (long long)N_STMT_ * 3);
  k_fill<<<grid1((long long)N_STMT_ * 3), 256, 0, stream>>>(nsum, 0.f, (long long)N_STMT_ * 3);
  {
    long long tot = (long long)E_STMT_ * 3;
    k_rgat_logits<<<grid1(tot), 256, 0, stream>>>(stmt_src, stmt_dst, stmt_et, h0s, h1s, rgat0_q, rgat0_k, elog, nmax, E_STMT_, 3, 384);
    k_edge_expsum<<<grid1(tot), 256, 0, stream>>>(stmt_dst, elog, nmax, nsum, E_STMT_, 0, 3);
    k_init_bias<<<grid1((long long)N_STMT_ * 384), 256, 0, stream>>>(outs, rgat0_b, N_STMT_, 384);
    k_rgat_scatter<<<grid1(tot), 256, 0, stream>>>(stmt_src, stmt_dst, stmt_et, elog, nsum, h0s, h1s, outs, E_STMT_, 3, 128);
  }
  k_bn_stats<<<384, 256, 0, stream>>>(outs, bnmean, bnvar, N_STMT_, 384, 1);
  k_bn_apply<<<grid1((long long)N_STMT_ * 384), 256, 0, stream>>>(outs, bnmean, bnvar, bnc0_g, bnc0_b, hbn, N_STMT_, 384, 1);

  // RGAT1
  gemm(hbn, rgat1_W,             nullptr, h0b, N_STMT_, 128, 384, 0);
  gemm(hbn, rgat1_W + 384 * 128, nullptr, h1b, N_STMT_, 128, 384, 0);
  k_fill<<<grid1(N_STMT_), 256, 0, stream>>>(nmax, -3.0e38f, N_STMT_);
  k_fill<<<grid1(N_STMT_), 256, 0, stream>>>(nsum, 0.f, N_STMT_);
  {
    long long tot = E_STMT_;
    k_rgat_logits<<<grid1(tot), 256, 0, stream>>>(stmt_src, stmt_dst, stmt_et, h0b, h1b, rgat1_q, rgat1_k, elog, nmax, E_STMT_, 1, 128);
    k_edge_expsum<<<grid1(tot), 256, 0, stream>>>(stmt_dst, elog, nmax, nsum, E_STMT_, 0, 1);
    k_init_bias<<<grid1((long long)N_STMT_ * 128), 256, 0, stream>>>(out2, rgat1_b, N_STMT_, 128);
    k_rgat_scatter<<<grid1(tot), 256, 0, stream>>>(stmt_src, stmt_dst, stmt_et, elog, nsum, h0b, h1b, out2, E_STMT_, 1, 128);
  }
  k_bn_stats<<<128, 256, 0, stream>>>(out2, bnmean, bnvar, N_STMT_, 128, 1);
  k_bn_apply<<<grid1((long long)N_STMT_ * 128), 256, 0, stream>>>(out2, bnmean, bnvar, bnc1_g, bnc1_b, hf, N_STMT_, 128, 1);

  // select + classifier head
  k_gather<<<grid1((long long)N_SEL_ * 128), 256, 0, stream>>>(hf, sel_idx, hsel, N_SEL_, 128);
  gemm(hsel, clf_w1, clf_b1, c1, N_SEL_, 128, 128, 1);
  k_clf_final<<<grid1(N_SEL_), 256, 0, stream>>>(c1, clf_w2, clf_b2, out, N_SEL_, 128);
}
